// CausalPointNetEncoder_48679159333441
// MI455X (gfx1250) — matrix operand order repaired and checked by reading, not compiled, not run on hardware
//
#include <hip/hip_runtime.h>
#include <hip/hip_bf16.h>
#include <cstdint>

typedef __attribute__((ext_vector_type(16))) _Float16 v16h;
typedef __attribute__((ext_vector_type(8)))  _Float16 v8h;
typedef __attribute__((ext_vector_type(8)))  float    v8f;

static constexpr int   NROWS = 16 * 64 * 256;   // 262144 rows (B*A*T)
static constexpr int   SEQ   = 16 * 64;         // 1024 sequences
static constexpr int   TLEN  = 256;
static constexpr int   CIN   = 32;
static constexpr int   HDIM  = 128;
static constexpr int   ODIM  = 64;
static constexpr int   MBLK  = 64;              // rows per block in GEMM
static constexpr int   APITCH = 40;             // LDS row pitch in halfs (80B, 16B-aligned, conflict-free)
static constexpr float EPSV  = 1e-5f;

// ---------------------------------------------------------------- utilities

__global__ void zero_f32(float* p, int n) {
    int i = blockIdx.x * blockDim.x + threadIdx.x;
    if (i < n) p[i] = 0.0f;
}

__global__ void cvt_f16(const float* __restrict__ in, _Float16* __restrict__ out, int n) {
    int i = blockIdx.x * blockDim.x + threadIdx.x;
    if (i < n) out[i] = (_Float16)in[i];
}

// W (K x Ncol, row major, fp32) -> Wt (Ncol x K, f16)  [column-major weights]
__global__ void transpose_cvt(const float* __restrict__ W, _Float16* __restrict__ Wt,
                              int K, int Ncol) {
    int i = blockIdx.x * blockDim.x + threadIdx.x;
    if (i < K * Ncol) {
        int k = i / Ncol, n = i % Ncol;
        Wt[(size_t)n * K + k] = (_Float16)W[i];
    }
}

// fold batch stats + gamma/beta into per-feature scale/shift
__global__ void bn_stats(const float* __restrict__ sum, const float* __restrict__ sumsq,
                         const float* __restrict__ g, const float* __restrict__ be,
                         float* __restrict__ scale, float* __restrict__ shift,
                         int nfeat, float invN) {
    int f = threadIdx.x;
    if (f < nfeat) {
        float mu  = sum[f] * invN;
        float var = sumsq[f] * invN - mu * mu;
        float s   = g[f] * rsqrtf(var + EPSV);
        scale[f] = s;
        shift[f] = be[f] - mu * s;
    }
}

// ---------------------------------------------------------------- WMMA GEMM
// Y[N x NCOL] = act(A[N x K]) @ W[K x NCOL] + bias ; W given transposed (NCOL x K) f16.
// A tiles staged global->LDS with double-buffered async copies (ASYNCcnt).
// K, NCOL, NT compile-time: copy loops fully unroll, K loop fully unrolls.
template <int K, int NCOL, int NT, bool BN_A, bool STATS, bool OUT_F32>
__global__ __launch_bounds__(NT)
void gemm_wmma(const _Float16* __restrict__ Amat,
               const _Float16* __restrict__ Bt,
               const float*    __restrict__ bias,
               void*           __restrict__ Yout,
               float* __restrict__ sumAcc, float* __restrict__ sqAcc,
               const float* __restrict__ aScale, const float* __restrict__ aShift) {
    __shared__ _Float16 Asmem[2][MBLK * APITCH];   // 2 x 5KB double buffer

    const int lane = threadIdx.x & 31;
    const int wave = threadIdx.x >> 5;       // one wave per 16-column slice
    const int hi   = lane >> 4;              // half-wave select
    const int l15  = lane & 15;
    const int n    = wave * 16 + l15;        // this lane's output column
    const int row0 = blockIdx.x * MBLK;

    const int akb = hi * 8;                  // A K-sub-offset  (ISA 16-bit A layout)
    const int bkb = hi * 16;                 // unused: B loads use akb (same k order as A)

    // precompute async-copy addresses: 256 x 16B chunks per 64x32h tile
    constexpr int CHUNKS = 256 / NT;         // 1 (NT=256) or 2 (NT=128)
    uint64_t gbase[CHUNKS];
    uint32_t lbase[2][CHUNKS];
#pragma unroll
    for (int i = 0; i < CHUNKS; ++i) {
        int c = threadIdx.x + i * NT;
        int row = c >> 2, chunk = c & 3;
        gbase[i]    = (uint64_t)(uintptr_t)(Amat + (size_t)(row0 + row) * K + chunk * 8);
        lbase[0][i] = (uint32_t)(uintptr_t)(&Asmem[0][row * APITCH + chunk * 8]);
        lbase[1][i] = (uint32_t)(uintptr_t)(&Asmem[1][row * APITCH + chunk * 8]);
    }

    v8f acc[4] = {v8f{}, v8f{}, v8f{}, v8f{}};

    // prologue: stage first K-step tile into buffer 0
#pragma unroll
    for (int i = 0; i < CHUNKS; ++i)
        asm volatile("global_load_async_to_lds_b128 %0, %1, off"
                     :: "v"(lbase[0][i]), "v"(gbase[i]) : "memory");

#pragma unroll
    for (int kc = 0; kc < K; kc += 32) {
        const int cur = (kc >> 5) & 1;
        asm volatile("s_wait_asynccnt 0x0" ::: "memory");
        __syncthreads();                         // tile[cur] visible to all waves

        if (kc + 32 < K) {                       // prefetch next tile into other buffer
#pragma unroll
            for (int i = 0; i < CHUNKS; ++i) {
                uint64_t g = gbase[i] + (uint64_t)((kc + 32) * 2);
                asm volatile("global_load_async_to_lds_b128 %0, %1, off"
                             :: "v"(lbase[cur ^ 1][i]), "v"(g) : "memory");
            }
        }

        // B tile: lane holds W^T[n][kc+akb .. +8) and W^T[n][kc+16+akb .. +8) (L2-resident)
        v16h b;
        {
            v8h b0 = *(const v8h*)(Bt + (size_t)n * K + kc + akb);
            v8h b1 = *(const v8h*)(Bt + (size_t)n * K + kc + 16 + akb);
#pragma unroll
            for (int j = 0; j < 8; ++j) { b[j] = b0[j]; b[8 + j] = b1[j]; }
        }

        v8f s0, s1, h0, h1;
        if (BN_A) {
            s0 = *(const v8f*)(aScale + kc + akb);
            s1 = *(const v8f*)(aScale + kc + 16 + akb);
            h0 = *(const v8f*)(aShift + kc + akb);
            h1 = *(const v8f*)(aShift + kc + 16 + akb);
        }

        // issue all LDS loads for the 4 row tiles before any WMMA
        v8h lo[4], hi8[4];
#pragma unroll
        for (int t = 0; t < 4; ++t) {
            const _Float16* ar = &Asmem[cur][(t * 16 + l15) * APITCH];
            lo[t]  = *(const v8h*)(ar + akb);
            hi8[t] = *(const v8h*)(ar + 16 + akb);
        }
#pragma unroll
        for (int t = 0; t < 4; ++t) {
            v16h a;
            if (BN_A) {
#pragma unroll
                for (int j = 0; j < 8; ++j) {
                    float v0 = fmaxf((float)lo[t][j]  * s0[j] + h0[j], 0.0f);
                    float v1 = fmaxf((float)hi8[t][j] * s1[j] + h1[j], 0.0f);
                    a[j]     = (_Float16)v0;
                    a[8 + j] = (_Float16)v1;
                }
            } else {
#pragma unroll
                for (int j = 0; j < 8; ++j) { a[j] = lo[t][j]; a[8 + j] = hi8[t][j]; }
            }
            acc[t] = __builtin_amdgcn_wmma_f32_16x16x32_f16(
                false, a, false, b, (short)0, acc[t], false, false);
        }
    }

    // bias + batch-stat partials
    const float bv = bias[n];
    float psum = 0.0f, psq = 0.0f;
#pragma unroll
    for (int t = 0; t < 4; ++t) {
#pragma unroll
        for (int r = 0; r < 8; ++r) {
            float v = acc[t][r] + bv;
            acc[t][r] = v;
            if (STATS) { psum += v; psq += v * v; }
        }
    }
    if (STATS) {
        psum += __shfl_xor(psum, 16, 32);   // combine the two row-halves of each column
        psq  += __shfl_xor(psq, 16, 32);
        if (hi == 0) {
            atomicAdd(&sumAcc[n], psum);
            atomicAdd(&sqAcc[n], psq);
        }
    }

    // store (C layout: lane -> col n ; VGPR r -> row r + 8*hi of each 16-row tile)
#pragma unroll
    for (int t = 0; t < 4; ++t) {
        int rowb = row0 + t * 16 + hi * 8;
#pragma unroll
        for (int r = 0; r < 8; ++r) {
            size_t idx = (size_t)(rowb + r) * NCOL + n;
            if (OUT_F32) ((float*)Yout)[idx] = acc[t][r];
            else         ((_Float16*)Yout)[idx] = (_Float16)acc[t][r];
        }
    }
}

// ------------------------------------------------- BN + ReLU + causal cummax
// layer 0: write concat [h, cummax(h)] rows of width 2H
__global__ void bnrelu_concat_cummax(const _Float16* __restrict__ y,
                                     const float* __restrict__ scale,
                                     const float* __restrict__ shift,
                                     _Float16* __restrict__ out) {
    int f   = threadIdx.x;     // 0..127
    int seq = blockIdx.x;      // 0..1023
    const _Float16* yp = y + (size_t)seq * TLEN * HDIM + f;
    _Float16*       op = out + (size_t)seq * TLEN * (2 * HDIM) + f;
    float s = scale[f], sh = shift[f];
    float run = 0.0f;
    for (int t = 0; t < TLEN; ++t) {
        float v = fmaxf((float)yp[(size_t)t * HDIM] * s + sh, 0.0f);
        run = fmaxf(run, v);
        op[(size_t)t * 2 * HDIM]        = (_Float16)v;
        op[(size_t)t * 2 * HDIM + HDIM] = (_Float16)run;
    }
}

// layer 2->3: only the cummax result survives
__global__ void bnrelu_cummax(const _Float16* __restrict__ y,
                              const float* __restrict__ scale,
                              const float* __restrict__ shift,
                              _Float16* __restrict__ out) {
    int f   = threadIdx.x;
    int seq = blockIdx.x;
    const _Float16* yp = y + (size_t)seq * TLEN * HDIM + f;
    _Float16*       op = out + (size_t)seq * TLEN * HDIM + f;
    float s = scale[f], sh = shift[f];
    float run = 0.0f;
    for (int t = 0; t < TLEN; ++t) {
        float v = fmaxf((float)yp[(size_t)t * HDIM] * s + sh, 0.0f);
        run = fmaxf(run, v);
        op[(size_t)t * HDIM] = (_Float16)run;
    }
}

// ---------------------------------------------------------------- launcher

extern "C" void kernel_launch(void* const* d_in, const int* in_sizes, int n_in,
                              void* d_out, int out_size, void* d_ws, size_t ws_size,
                              hipStream_t stream) {
    const float* poly = (const float*)d_in[0];
    const float* W0 = (const float*)d_in[1];  const float* b0 = (const float*)d_in[2];
    const float* g0 = (const float*)d_in[3];  const float* be0 = (const float*)d_in[4];
    const float* W1 = (const float*)d_in[5];  const float* b1 = (const float*)d_in[6];
    const float* g1 = (const float*)d_in[7];  const float* be1 = (const float*)d_in[8];
    const float* W2 = (const float*)d_in[9];  const float* b2 = (const float*)d_in[10];
    const float* g2 = (const float*)d_in[11]; const float* be2 = (const float*)d_in[12];
    const float* W3 = (const float*)d_in[13]; const float* b3 = (const float*)d_in[14];
    const float* g3 = (const float*)d_in[15]; const float* be3 = (const float*)d_in[16];
    const float* W4 = (const float*)d_in[17]; const float* b4 = (const float*)d_in[18];
    float* out = (float*)d_out;

    // ---- workspace layout
    char*  ws    = (char*)d_ws;
    float* accum = (float*)ws;              // 4 stages * (128 sum + 128 sumsq)
    float* bnp   = accum + 1024;            // 4 stages * (128 scale + 128 shift)
    _Float16* W0t = (_Float16*)(bnp + 1024);
    _Float16* W1t = W0t + 128 * 32;
    _Float16* W2t = W1t + 128 * 256;
    _Float16* W3t = W2t + 128 * 128;
    _Float16* W4t = W3t + 128 * 128;        // 64 x 128
    char* actbase = ws + (256 << 10);
    _Float16* actA = (_Float16*)actbase;                 // 128MB : x1 (N x 256) -> later y2 (N x 128)
    _Float16* actB = actA + (size_t)NROWS * 256;         //  64MB : y0 -> y1 -> y3
    _Float16* actC = actB + (size_t)NROWS * 128;         //  64MB : x3
    _Float16* actD = actC + (size_t)NROWS * 128;         //  17MB : input f16

    float* sum0 = accum;        float* sq0 = sum0 + 128;
    float* sum1 = accum + 256;  float* sq1 = sum1 + 128;
    float* sum2 = accum + 512;  float* sq2 = sum2 + 128;
    float* sum3 = accum + 768;  float* sq3 = sum3 + 128;
    float* sc0 = bnp;        float* sh0 = sc0 + 128;
    float* sc1 = bnp + 256;  float* sh1 = sc1 + 128;
    float* sc2 = bnp + 512;  float* sh2 = sc2 + 128;
    float* sc3 = bnp + 768;  float* sh3 = sc3 + 128;

    const float invN = 1.0f / (float)NROWS;
    const int gemmGrid = NROWS / MBLK;      // 4096

    // ---- prep
    zero_f32<<<4, 256, 0, stream>>>(accum, 1024);
    transpose_cvt<<<16, 256, 0, stream>>>(W0, W0t, CIN, HDIM);
    transpose_cvt<<<128, 256, 0, stream>>>(W1, W1t, 2 * HDIM, HDIM);
    transpose_cvt<<<64, 256, 0, stream>>>(W2, W2t, HDIM, HDIM);
    transpose_cvt<<<64, 256, 0, stream>>>(W3, W3t, HDIM, HDIM);
    transpose_cvt<<<32, 256, 0, stream>>>(W4, W4t, HDIM, ODIM);
    cvt_f16<<<(NROWS * CIN) / 256, 256, 0, stream>>>(poly, actD, NROWS * CIN);

    // ---- layer 0: y0 = x @ W0 + b0 (+stats)
    gemm_wmma<CIN, HDIM, 256, false, true, false><<<gemmGrid, 256, 0, stream>>>(
        actD, W0t, b0, actB, sum0, sq0, nullptr, nullptr);
    bn_stats<<<1, 128, 0, stream>>>(sum0, sq0, g0, be0, sc0, sh0, HDIM, invN);

    // ---- BN0+ReLU, concat with causal cummax -> x1 (N x 256)
    bnrelu_concat_cummax<<<SEQ, HDIM, 0, stream>>>(actB, sc0, sh0, actA);

    // ---- layer 1: y1 = x1 @ W1 + b1 (+stats)
    gemm_wmma<2 * HDIM, HDIM, 256, false, true, false><<<gemmGrid, 256, 0, stream>>>(
        actA, W1t, b1, actB, sum1, sq1, nullptr, nullptr);
    bn_stats<<<1, 128, 0, stream>>>(sum1, sq1, g1, be1, sc1, sh1, HDIM, invN);

    // ---- layer 2: y2 = bn1relu(y1) @ W2 + b2 (+stats)   [BN fused into A load]
    gemm_wmma<HDIM, HDIM, 256, true, true, false><<<gemmGrid, 256, 0, stream>>>(
        actB, W2t, b2, actA, sum2, sq2, sc1, sh1);
    bn_stats<<<1, 128, 0, stream>>>(sum2, sq2, g2, be2, sc2, sh2, HDIM, invN);

    // ---- BN2+ReLU + causal cummax -> x3
    bnrelu_cummax<<<SEQ, HDIM, 0, stream>>>(actA, sc2, sh2, actC);

    // ---- layer 3: y3 = x3 @ W3 + b3 (+stats)
    gemm_wmma<HDIM, HDIM, 256, false, true, false><<<gemmGrid, 256, 0, stream>>>(
        actC, W3t, b3, actB, sum3, sq3, nullptr, nullptr);
    bn_stats<<<1, 128, 0, stream>>>(sum3, sq3, g3, be3, sc3, sh3, HDIM, invN);

    // ---- layer 4: out = bn3relu(y3) @ W4 + b4  (fp32 out, BN fused)
    gemm_wmma<HDIM, ODIM, 128, true, false, true><<<gemmGrid, 128, 0, stream>>>(
        actB, W4t, b4, out, nullptr, nullptr, sc3, sh3);
}
